// LSTMAutoEncoder_29721173688881
// MI455X (gfx1250) — matrix operand order repaired and checked by reading, not compiled, not run on hardware
//
#include <hip/hip_runtime.h>
#include <hip/hip_bf16.h>

// LSTM autoencoder: 8 layers, B=128, T=512, all dims 256, gate dim 1024.
// Phase 1 per layer: xp = x·Wihᵀ + b   (WMMA GEMM, xp stored in C-fragment layout)
// Phase 2 per layer: sequential scan. Per step:
//   - next step's xp slab staged LDS-ward via global_load_async_to_lds_b128 (ASYNCcnt)
//   - Whh B-fragments double-buffered in VGPRs across k-steps (L2 latency hidden)
//   - h double-buffered in LDS (f16), c in VGPRs (f32), activations via v_tanh_f32

#define TS 512
#define BS 128
#define HD 256
#define GD 1024
#define NLAYER 8

typedef __attribute__((ext_vector_type(16))) _Float16 v16h;
typedef __attribute__((ext_vector_type(8)))  _Float16 v8h;
typedef __attribute__((ext_vector_type(8)))  float    v8f;

__device__ __forceinline__ float fast_tanh(float x) {
  float r;
  // TRANS op: 1 independent op required before result use (ISA 7.4.3) -> v_nop inside asm
  asm("v_tanh_f32 %0, %1\n\tv_nop" : "=v"(r) : "v"(x));
  return r;
}
__device__ __forceinline__ float fsig(float x) { return 0.5f * fast_tanh(0.5f * x) + 0.5f; }

// A-fragment (16x32 f16, MxK) from row-major, per ISA layout.
__device__ __forceinline__ v16h load_a_frag(const _Float16* __restrict__ tile, int ld, int lane) {
  int row = lane & 15;
  int kb  = (lane >> 4) << 3;
  const _Float16* p = tile + row * ld + kb;
  v8h lo = *(const v8h*)p;
  v8h hi = *(const v8h*)(p + 16);
  v16h r;
#pragma unroll
  for (int i = 0; i < 8; ++i) { r[i] = lo[i]; r[i + 8] = hi[i]; }
  return r;
}

// B-fragment (32x16 f16, KxN), B[k,n]=W[n,k], W row-major ld=HD: two 16B loads/lane (k=8h.., 16+8h..).
__device__ __forceinline__ v16h load_b_frag(const _Float16* __restrict__ Wtile, int lane) {
  int col = lane & 15;
  int kb  = (lane >> 4) << 3;
  const _Float16* p = Wtile + col * HD + kb;
  v8h lo = *(const v8h*)p;
  v8h hi = *(const v8h*)(p + 16);
  v16h r;
#pragma unroll
  for (int i = 0; i < 8; ++i) { r[i] = lo[i]; r[i + 8] = hi[i]; }
  return r;
}

// ---------- transpose+convert layer-0 input: x[b,t,d] f32 -> xT[t,b,d] f16 ----------
__global__ void __launch_bounds__(256) conv_x_kernel(const float* __restrict__ x,
                                                     _Float16* __restrict__ xT) {
  size_t idx = (size_t)blockIdx.x * 256 + threadIdx.x;   // over T*B*256
  int d = (int)(idx & 255);
  size_t bt = idx >> 8;
  int b = (int)(bt & 127);
  int t = (int)(bt >> 7);
  xT[idx] = (_Float16)x[((size_t)b * TS + t) * HD + d];
}

// ---------- weight convert f32->f16 + bias fuse ----------
__global__ void __launch_bounds__(256) conv_w_kernel(const float* __restrict__ Wih,
                                                     const float* __restrict__ Whh,
                                                     const float* __restrict__ bih,
                                                     const float* __restrict__ bhh,
                                                     _Float16* __restrict__ wih16,
                                                     _Float16* __restrict__ whh16,
                                                     float* __restrict__ bias) {
  int idx = blockIdx.x * 256 + threadIdx.x;              // 0 .. GD*HD-1
  wih16[idx] = (_Float16)Wih[idx];
  whh16[idx] = (_Float16)Whh[idx];
  if (idx < GD) bias[idx] = bih[idx] + bhh[idx];
}

// ---------- input projection GEMM ----------
// rows r = t*B+b (65536); 1 wave = 16x64 tile (1 M-tile x 4 N-tiles).
// Output fragment layout: xpF[((mt*64 + ntile)*32 + lane)*8 + v], mt = t*8 + chunk.
__global__ void __launch_bounds__(256) lstm_in_gemm(const _Float16* __restrict__ xT,
                                                    const _Float16* __restrict__ Wih,
                                                    const float* __restrict__ bias,
                                                    float* __restrict__ xpF) {
  const int lane = threadIdx.x & 31;
  const int wave = threadIdx.x >> 5;
  const int job  = blockIdx.x * 8 + wave;   // 65536 jobs
  const int mt   = job >> 4;                // 0..4095 (M tile = t*8 + chunk)
  const int ng   = job & 15;                // N group of 4 tiles
  const int m0   = mt * 16;
  const int n0   = ng * 64;

  v8f acc[4];
#pragma unroll
  for (int j = 0; j < 4; ++j) acc[j] = (v8f){0.f,0.f,0.f,0.f,0.f,0.f,0.f,0.f};

  const _Float16* Abase = xT + (size_t)m0 * HD;
#pragma unroll
  for (int k0 = 0; k0 < HD; k0 += 32) {
    v16h a = load_a_frag(Abase + k0, HD, lane);
#pragma unroll
    for (int j = 0; j < 4; ++j) {
      v16h b = load_b_frag(Wih + (size_t)(n0 + j * 16) * HD + k0, lane);
      acc[j] = __builtin_amdgcn_wmma_f32_16x16x32_f16(false, a, false, b,
                                                      (short)0, acc[j], false, false);
    }
  }

  const int coll = lane & 15;
#pragma unroll
  for (int j = 0; j < 4; ++j) {
    float bsum = bias[n0 + j * 16 + coll];
    v8f outv;
#pragma unroll
    for (int v = 0; v < 8; ++v) outv[v] = acc[j][v] + bsum;
    *(v8f*)&xpF[((size_t)(mt * 64 + ng * 4 + j) * 32 + lane) * 8] = outv;
  }
}

// async stage: one thread copies 16B; same IOFFSET applies to both LDS dst and global src
#define ASYNC_CP(J) \
  asm volatile("global_load_async_to_lds_b128 %0, %1, %2 offset:" #J \
               :: "v"(ldsdst), "v"(voff), "s"(gbase) : "memory")
#define ASYNC_STAGE_ALL() do { \
  ASYNC_CP(0);     ASYNC_CP(4096);  ASYNC_CP(8192);  ASYNC_CP(12288); \
  ASYNC_CP(16384); ASYNC_CP(20480); ASYNC_CP(24576); ASYNC_CP(28672); \
  ASYNC_CP(32768); ASYNC_CP(36864); ASYNC_CP(40960); ASYNC_CP(45056); \
  ASYNC_CP(49152); ASYNC_CP(53248); ASYNC_CP(57344); ASYNC_CP(61440); } while (0)

// ---------- recurrent scan ----------
// 8 blocks x 16 batch rows; 8 waves x 32 hidden cols.
__global__ void __launch_bounds__(256) lstm_scan(const float* __restrict__ xpF,     // fragment layout
                                                 const _Float16* __restrict__ Whh,  // [GD,HD]
                                                 _Float16* __restrict__ xT,         // [T,B,HD] next-layer input
                                                 float* __restrict__ outF) {        // [B,T,HD] or null
  __shared__ __align__(16) _Float16 hbuf[2][16 * HD];   // 2 x 8 KB
  __shared__ __align__(16) float xstage[2][16 * GD];    // 2 x 64 KB

  const int tid   = threadIdx.x;
  const int lane  = tid & 31;
  const int wave  = tid >> 5;
  const int chunk = blockIdx.x;       // batch chunk (16 rows)
  const int b0    = chunk * 16;
  const int hc0   = wave * 32;        // hidden-column slice owned by this wave

  const unsigned voff  = (unsigned)tid * 16;                               // global byte offset
  const unsigned ldsa0 = (unsigned)(uintptr_t)&xstage[0][0] + (unsigned)tid * 16;
  const unsigned ldsa1 = (unsigned)(uintptr_t)&xstage[1][0] + (unsigned)tid * 16;

  for (int i = tid; i < 16 * HD; i += 256) hbuf[0][i] = (_Float16)0.0f;

  // prologue: stage xp slab for t=0 into xstage[0]
  {
    unsigned ldsdst = ldsa0;
    unsigned long long gbase = (unsigned long long)(uintptr_t)(xpF + (size_t)chunk * 64 * 256);
    ASYNC_STAGE_ALL();
    asm volatile("s_wait_asynccnt 0x0" ::: "memory");
  }
  __syncthreads();

  const int row_hi = (lane >> 4) << 3;
  const int coll   = lane & 15;

  // Whh tile bases for this wave's 8 (q,s) tiles, il = q*2+s
  const _Float16* wbase[8];
#pragma unroll
  for (int il = 0; il < 8; ++il)
    wbase[il] = Whh + (size_t)((il >> 1) * 256 + hc0 + (il & 1) * 16) * HD;

  // prologue: preload k0=0 B-fragments
  v16h bb[2][8];
#pragma unroll
  for (int il = 0; il < 8; ++il) bb[0][il] = load_b_frag(wbase[il], lane);

  v8f cst[2];
#pragma unroll
  for (int s = 0; s < 2; ++s) cst[s] = (v8f){0.f,0.f,0.f,0.f,0.f,0.f,0.f,0.f};

  for (int t = 0; t < TS; ++t) {
    const int rb = t & 1;        // h read buffer / xstage read buffer
    const int wb = rb ^ 1;       // h write buffer / xstage write buffer

    // 1) stage next step's xp slab (full step of latency cover)
    if (t + 1 < TS) {
      unsigned ldsdst = wb ? ldsa1 : ldsa0;
      unsigned long long gbase =
          (unsigned long long)(uintptr_t)(xpF + ((size_t)(t + 1) * 8 + chunk) * 64 * 256);
      ASYNC_STAGE_ALL();
    }

    // 2) gates = h · Whhᵀ; B-fragments double-buffered across k-steps
    v8f acc[8];
#pragma unroll
    for (int il = 0; il < 8; ++il) acc[il] = (v8f){0.f,0.f,0.f,0.f,0.f,0.f,0.f,0.f};

#pragma unroll
    for (int kk = 0; kk < 8; ++kk) {
      if (kk < 7) {
#pragma unroll
        for (int il = 0; il < 8; ++il)
          bb[(kk + 1) & 1][il] = load_b_frag(wbase[il] + (kk + 1) * 32, lane);
      }
      int arow = lane & 15;
      int akb  = kk * 32 + ((lane >> 4) << 3);
      v8h lo = *(const v8h*)&hbuf[rb][arow * HD + akb];
      v8h hi = *(const v8h*)&hbuf[rb][arow * HD + akb + 16];
      v16h a;
#pragma unroll
      for (int i = 0; i < 8; ++i) { a[i] = lo[i]; a[i + 8] = hi[i]; }
#pragma unroll
      for (int il = 0; il < 8; ++il)
        acc[il] = __builtin_amdgcn_wmma_f32_16x16x32_f16(false, a, false, bb[kk & 1][il],
                                                         (short)0, acc[il], false, false);
    }
    // re-issue k0=0 fragments for next step (t-invariant addresses; overlaps tail)
#pragma unroll
    for (int il = 0; il < 8; ++il) bb[0][il] = load_b_frag(wbase[il], lane);

    // 3) add staged xp, activations, state update; write h to OTHER LDS buffer
    const float* xs = &xstage[rb][0];
#pragma unroll
    for (int s = 0; s < 2; ++s) {
      v8f xq[4];
#pragma unroll
      for (int q = 0; q < 4; ++q) {
        int ntile = q * 16 + wave * 2 + s;
        xq[q] = *(const v8f*)&xs[ntile * 256 + lane * 8];
      }
#pragma unroll
      for (int v = 0; v < 8; ++v) {
        float iv = fsig(acc[0 * 2 + s][v] + xq[0][v]);
        float fv = fsig(acc[1 * 2 + s][v] + xq[1][v]);
        float gv = fast_tanh(acc[2 * 2 + s][v] + xq[2][v]);
        float ov = fsig(acc[3 * 2 + s][v] + xq[3][v]);
        float c  = fv * cst[s][v] + iv * gv;
        cst[s][v] = c;
        float h  = ov * fast_tanh(c);
        int row = v + row_hi;            // 0..15 within batch chunk
        int col = hc0 + s * 16 + coll;   // 0..255 hidden col
        hbuf[wb][row * HD + col] = (_Float16)h;
        if (outF) outF[((size_t)(b0 + row) * TS + t) * HD + col] = h;
      }
    }

    // 4) wait for next-step staging, then one barrier publishes h(t) and xstage(t+1)
    asm volatile("s_wait_asynccnt 0x0" ::: "memory");
    __syncthreads();

    // 5) coalesced LDS -> global copy of h(t) for the next layer's GEMM input.
    {
      const int4* src = (const int4*)&hbuf[wb][0];
      int4* dst = (int4*)(xT + ((size_t)t * BS + b0) * HD);
      dst[tid * 2]     = src[tid * 2];
      dst[tid * 2 + 1] = src[tid * 2 + 1];
    }
  }
}

extern "C" void kernel_launch(void* const* d_in, const int* in_sizes, int n_in,
                              void* d_out, int out_size, void* d_ws, size_t ws_size,
                              hipStream_t stream) {
  (void)in_sizes; (void)n_in; (void)out_size; (void)ws_size;
  const float* x = (const float*)d_in[0];

  // workspace carve-up (~310 MB): xT f16 | xpF f32 | wih16 | whh16 | bias
  char* ws = (char*)d_ws;
  size_t off = 0;
  _Float16* xT    = (_Float16*)(ws + off); off += (size_t)TS * BS * HD * sizeof(_Float16);
  float*    xpF   = (float*)   (ws + off); off += (size_t)TS * BS * GD * sizeof(float);
  _Float16* wih16 = (_Float16*)(ws + off); off += (size_t)NLAYER * GD * HD * sizeof(_Float16);
  _Float16* whh16 = (_Float16*)(ws + off); off += (size_t)NLAYER * GD * HD * sizeof(_Float16);
  float*    bias  = (float*)   (ws + off); off += (size_t)NLAYER * GD * sizeof(float);

  conv_x_kernel<<<TS * BS, 256, 0, stream>>>(x, xT);

  for (int l = 0; l < NLAYER; ++l) {
    const float* Wih = (const float*)d_in[1 + l * 4 + 0];
    const float* Whh = (const float*)d_in[1 + l * 4 + 1];
    const float* bih = (const float*)d_in[1 + l * 4 + 2];
    const float* bhh = (const float*)d_in[1 + l * 4 + 3];
    conv_w_kernel<<<(GD * HD) / 256, 256, 0, stream>>>(
        Wih, Whh, bih, bhh,
        wih16 + (size_t)l * GD * HD, whh16 + (size_t)l * GD * HD, bias + (size_t)l * GD);
  }

  for (int l = 0; l < NLAYER; ++l) {
    lstm_in_gemm<<<8192, 256, 0, stream>>>(xT, wih16 + (size_t)l * GD * HD,
                                           bias + (size_t)l * GD, xpF);
    lstm_scan<<<8, 256, 0, stream>>>(xpF, whh16 + (size_t)l * GD * HD, xT,
                                     (l == NLAYER - 1) ? (float*)d_out : nullptr);
  }
}
